// InterestEvolve_37469294690961
// MI455X (gfx1250) — matrix operand order repaired and checked by reading, not compiled, not run on hardware
//
#include <hip/hip_runtime.h>

// ---------------------------------------------------------------------------
// Attention-gated GRU scan for MI455X (gfx1250, wave32, WMMA).
// B=2048, T=200, D=128, U=128.  One block = 16 batch rows, resident for all T.
//  - weights transposed (f16) in the 320KB WGP LDS (~203 KB)
//  - hidden state h: fp32, entirely in registers (8 per lane, tiles 16x128)
//  - x_{t+1} / a_{t+1} global loads software-pipelined behind step t's WMMAs
//  - all GEMM work: v_wmma_f32_16x16x32_f16 (24 per wave per step)
// ---------------------------------------------------------------------------

typedef _Float16 v16h __attribute__((ext_vector_type(16)));
typedef _Float16 v8h  __attribute__((ext_vector_type(8)));
typedef float    v8f  __attribute__((ext_vector_type(8)));

#define T_LEN 200
#define D_DIM 128
#define U_DIM 128
#define K_TOT 256          // U + D
#define BM    16           // batch rows per block
#define NTHREADS 256       // 8 waves

// LDS row strides (halves), padded for bank-conflict avoidance and 16-byte
// alignment of every fragment load.
#define WSTR 264           // weight rows [n][k], 264 halves = 528 B
#define ASTR 264           // [h | x] concat buffer
#define RSTR 136           // r*h buffer (128 cols)

__device__ __forceinline__ float sigm(float x) {
    return 1.0f / (1.0f + __expf(-x));
}
__device__ __forceinline__ float tanh_fast(float x) {
    // 1 - 2/(e^{2x}+1): monotone, saturates correctly for |x| large
    return 1.0f - 2.0f / (__expf(2.0f * x) + 1.0f);
}

__device__ __forceinline__ v16h ld_frag16(const _Float16* p0, const _Float16* p1) {
    v8h a = *(const v8h*)p0;
    v8h b = *(const v8h*)p1;
    return __builtin_shufflevector(a, b, 0,1,2,3,4,5,6,7,8,9,10,11,12,13,14,15);
}

__global__ __launch_bounds__(NTHREADS, 1)
void gru_scan_wmma(const float* __restrict__ hist,   // (B,T,D)
                   const float* __restrict__ attn,   // (B,T,1)
                   const float* __restrict__ Wu, const float* __restrict__ bu,
                   const float* __restrict__ Wr, const float* __restrict__ br,
                   const float* __restrict__ Wh, const float* __restrict__ bh,
                   float* __restrict__ out)          // (B,U)
{
    // ---- LDS (~216 KB of the 320 KB WGP pool) ----
    __shared__ _Float16 sW[3][U_DIM][WSTR]; // transposed weights [gate][n][k]
    __shared__ _Float16 sA[BM][ASTR];       // A for u/r gates: [h(0:128) | x(128:256)]
    __shared__ _Float16 sR[BM][RSTR];       // r*h (f16), A-tail for candidate gate
    __shared__ float    sAtt[BM];           // attention score for current t

    const int tid  = threadIdx.x;
    const int lane = tid & 31;
    const int wave = tid >> 5;        // 0..7 : owns output cols N0..N0+15
    const int N0   = wave * 16;
    const int lo   = lane & 15;
    const int hi   = lane >> 4;       // 0/1 : lane-half for fragment layouts
    const int b0   = blockIdx.x * BM;

    // ---- one-time: load & transpose weights f32->f16 into LDS ----
    {
        const float* Wsrc[3] = { Wu, Wr, Wh };
        for (int g = 0; g < 3; ++g) {
            const float* W = Wsrc[g];
            for (int idx = tid; idx < K_TOT * U_DIM; idx += NTHREADS) {
                int k = idx >> 7;      // 0..255  (row of W)
                int n = idx & 127;     // 0..127  (col of W)
                sW[g][n][k] = (_Float16)W[idx];
            }
        }
        for (int idx = tid; idx < BM * U_DIM; idx += NTHREADS) {
            int r = idx >> 7, c = idx & 127;
            sA[r][c] = (_Float16)0.0f;       // h part of A buffer (h0 = 0)
        }
    }

    // Per-lane B-fragment bases: column N0+lo, k-base hi*8 (8 halves, then 8 more at +16)
    const _Float16* wu_b = &sW[0][N0 + lo][hi * 8];
    const _Float16* wr_b = &sW[1][N0 + lo][hi * 8];
    const _Float16* wh_b = &sW[2][N0 + lo][hi * 8];

    const float biasu = bu[N0 + lo];
    const float biasr = br[N0 + lo];
    const float biash = bh[N0 + lo];

    // Hidden state fp32 in registers: lane owns (row = i + 8*hi, col = N0+lo)
    float hreg[8];
    #pragma unroll
    for (int i = 0; i < 8; ++i) hreg[i] = 0.0f;

    // x-loader mapping: thread -> (row, 8 cols); software-pipelined prefetch
    const int xrow = tid >> 4;            // 0..15
    const int xcol = (tid & 15) * 8;      // 0..120
    const float* xbase = hist + (size_t)(b0 + xrow) * T_LEN * D_DIM + xcol;

    float4 x0 = *(const float4*)(xbase);
    float4 x1 = *(const float4*)(xbase + 4);
    float  av = (tid < BM) ? attn[(size_t)(b0 + tid) * T_LEN] : 0.0f;

    for (int t = 0; t < T_LEN; ++t) {
        // ===== phase 1: stage pre-fetched x_t (packed f16) and a_t =====
        {
            v8h xv;
            xv[0] = (_Float16)x0.x; xv[1] = (_Float16)x0.y;
            xv[2] = (_Float16)x0.z; xv[3] = (_Float16)x0.w;
            xv[4] = (_Float16)x1.x; xv[5] = (_Float16)x1.y;
            xv[6] = (_Float16)x1.z; xv[7] = (_Float16)x1.w;
            *(v8h*)&sA[xrow][128 + xcol] = xv;
        }
        if (tid < BM) sAtt[tid] = av;
        __syncthreads();

        // issue next step's global loads now; latency hidden by phases 2-5
        if (t + 1 < T_LEN) {
            const float* xp = xbase + (size_t)(t + 1) * D_DIM;
            x0 = *(const float4*)(xp);
            x1 = *(const float4*)(xp + 4);
            if (tid < BM) av = attn[(size_t)(b0 + tid) * T_LEN + (t + 1)];
        }

        // ===== phase 2: u_pre, r_pre = [h|x] @ {Wu,Wr}  (16 WMMAs) =====
        v8f cu, cr;
        #pragma unroll
        for (int i = 0; i < 8; ++i) { cu[i] = biasu; cr[i] = biasr; }

        #pragma unroll
        for (int kt = 0; kt < 8; ++kt) {
            const _Float16* ap = &sA[lo][kt * 32 + hi * 8];
            v16h af  = ld_frag16(ap, ap + 16);
            v16h bfu = ld_frag16(wu_b + kt * 32, wu_b + kt * 32 + 16);
            v16h bfr = ld_frag16(wr_b + kt * 32, wr_b + kt * 32 + 16);
            cu = __builtin_amdgcn_wmma_f32_16x16x32_f16(false, af, false, bfu,
                                                        (short)0, cu, false, false);
            cr = __builtin_amdgcn_wmma_f32_16x16x32_f16(false, af, false, bfr,
                                                        (short)0, cr, false, false);
        }

        // ===== phase 3: r = sigmoid(r_pre); stage r*h (h from registers) =====
        #pragma unroll
        for (int i = 0; i < 8; ++i) {
            int row = i + hi * 8;
            sR[row][N0 + lo] = (_Float16)(sigm(cr[i]) * hreg[i]);
        }
        __syncthreads();

        // ===== phase 4: hh_pre = [x | r*h] @ Wh  (8 WMMAs) =====
        v8f ch;
        #pragma unroll
        for (int i = 0; i < 8; ++i) ch[i] = biash;

        #pragma unroll
        for (int kt = 0; kt < 8; ++kt) {
            const _Float16* ap = (kt < 4)
                ? &sA[lo][128 + kt * 32 + hi * 8]          // x part (K 0..127)
                : &sR[lo][(kt - 4) * 32 + hi * 8];         // r*h part (K 128..255)
            v16h af  = ld_frag16(ap, ap + 16);
            v16h bfh = ld_frag16(wh_b + kt * 32, wh_b + kt * 32 + 16);
            ch = __builtin_amdgcn_wmma_f32_16x16x32_f16(false, af, false, bfh,
                                                        (short)0, ch, false, false);
        }

        // ===== phase 5: h = u*a*hh + (1-u*a)*h  (registers) =====
        #pragma unroll
        for (int i = 0; i < 8; ++i) {
            int row = i + hi * 8;
            float u    = sigm(cu[i]) * sAtt[row];
            float hh   = tanh_fast(ch[i]);
            float hnew = u * hh + (1.0f - u) * hreg[i];
            hreg[i] = hnew;
            sA[row][N0 + lo] = (_Float16)hnew;   // h part of next step's A
        }
        __syncthreads();
    }

    // ---- write final h (B,U) fp32 straight from registers ----
    #pragma unroll
    for (int i = 0; i < 8; ++i) {
        int row = i + hi * 8;
        out[(size_t)(b0 + row) * U_DIM + (N0 + lo)] = hreg[i];
    }
}

extern "C" void kernel_launch(void* const* d_in, const int* in_sizes, int n_in,
                              void* d_out, int out_size, void* d_ws, size_t ws_size,
                              hipStream_t stream) {
    (void)n_in; (void)out_size; (void)d_ws; (void)ws_size;
    const float* hist = (const float*)d_in[0];
    const float* attn = (const float*)d_in[1];
    const float* Wu   = (const float*)d_in[2];
    const float* bu   = (const float*)d_in[3];
    const float* Wr   = (const float*)d_in[4];
    const float* br   = (const float*)d_in[5];
    const float* Wh   = (const float*)d_in[6];
    const float* bh   = (const float*)d_in[7];
    float* out = (float*)d_out;

    const int B = in_sizes[0] / (T_LEN * D_DIM);   // 2048
    dim3 grid(B / BM), block(NTHREADS);
    gru_scan_wmma<<<grid, block, 0, stream>>>(hist, attn, Wu, bu, Wr, br, Wh, bh, out);
}
